// CosFormerLayer_42030549959124
// MI455X (gfx1250) — compile-verified, matrix operand order checked by reading, not run on hardware
//
#include <hip/hip_runtime.h>
#include <hip/hip_bf16.h>

// ---------------------------------------------------------------------------
// CosFormer layer, CDNA5 (gfx1250) wave32 + WMMA bf16 implementation.
// GEMMs: async-to-LDS double-buffered pipeline feeding v_wmma_f32_16x16x32_bf16.
// ---------------------------------------------------------------------------

typedef __attribute__((ext_vector_type(16))) __bf16 v16bf;
typedef __attribute__((ext_vector_type(8)))  float  v8f;

#define BB   32
#define SS   512
#define DD   512
#define HH   8
#define DKK  64
#define DVV  64
#define DFF_ 2048
#define BS_  (BB * SS)          // 16384 rows
#define QKVW (3 * HH * DKK)     // 1536

union FragBF { uint4 u[2]; v16bf v; };
union Pack8  { uint4 u; __bf16 h[8]; };

__device__ __forceinline__ v16bf lds_frag(const __bf16* p) {
  FragBF f;
  f.u[0] = *reinterpret_cast<const uint4*>(p);
  f.u[1] = *reinterpret_cast<const uint4*>(p + 16);
  return f.v;
}

__device__ __forceinline__ v8f wmma_bf16(v16bf a, v16bf b, v8f c) {
  return __builtin_amdgcn_wmma_f32_16x16x32_bf16(
      /*neg_a=*/false, a, /*neg_b=*/false, b,
      /*c_mod=*/(short)0, c, /*reuse_a=*/false, /*reuse_b=*/false);
}

// Async DMA of 16 bytes global -> LDS (per lane). Tracked by ASYNCcnt.
__device__ __forceinline__ void async_copy_b128(unsigned int lds_addr,
                                                const __bf16* gptr) {
  unsigned long long ga = (unsigned long long)(size_t)gptr;
  asm volatile("global_load_async_to_lds_b128 %0, %1, off"
               :: "v"(lds_addr), "v"(ga) : "memory");
}

// ---------------------------------------------------------------------------
// Elementwise f32 -> bf16 conversion
// ---------------------------------------------------------------------------
__global__ __launch_bounds__(256)
void cvt_bf16_kernel(const float* __restrict__ in, __bf16* __restrict__ out, int n4) {
  int t = blockIdx.x * blockDim.x + threadIdx.x;
  if (t >= n4) return;
  float4 v = reinterpret_cast<const float4*>(in)[t];
  __bf16* o = out + (size_t)t * 4;
  o[0] = (__bf16)v.x; o[1] = (__bf16)v.y; o[2] = (__bf16)v.z; o[3] = (__bf16)v.w;
}

// ---------------------------------------------------------------------------
// Transpose + convert: W[K x N] f32 -> Wt[N x K] bf16
// ---------------------------------------------------------------------------
__global__ __launch_bounds__(256)
void transpose_cvt_kernel(const float* __restrict__ W, __bf16* __restrict__ Wt,
                          int K, int N) {
  __shared__ float tile[32][33];
  int n0 = blockIdx.x * 32, k0 = blockIdx.y * 32;
  int tx = threadIdx.x & 31, ty = threadIdx.x >> 5;   // ty in 0..7
#pragma unroll
  for (int i = 0; i < 4; i++)
    tile[ty + i * 8][tx] = W[(size_t)(k0 + ty + i * 8) * N + n0 + tx];
  __syncthreads();
#pragma unroll
  for (int i = 0; i < 4; i++)
    Wt[(size_t)(n0 + ty + i * 8) * K + k0 + tx] = (__bf16)tile[tx][ty + i * 8];
}

// ---------------------------------------------------------------------------
// Generic bf16 WMMA GEMM:  C[M x N] = A[M x K] * Bt[N x K]^T (+bias)(relu)(+res)
// Block 256 threads = 8 waves, 128x128 tile, wave tile 64x32 (8 WMMA accs).
// K staged 32 wide via GLOBAL_LOAD_ASYNC_TO_LDS_B128, double-buffered LDS.
// ---------------------------------------------------------------------------
__global__ __launch_bounds__(256)
void gemm_bf16_kernel(const __bf16* __restrict__ A, const __bf16* __restrict__ Bt,
                      int M, int N, int K,
                      const float* __restrict__ bias,
                      const float* __restrict__ resid,
                      float* __restrict__ outF, __bf16* __restrict__ outB,
                      int relu)
{
  __shared__ __align__(16) __bf16 As[2][128 * 40];
  __shared__ __align__(16) __bf16 Bs[2][128 * 40];

  const int tid  = threadIdx.x;
  const int lane = tid & 31;
  const int wave = tid >> 5;
  const int wm = wave >> 2;          // 0..1
  const int wn = wave & 3;           // 0..3
  const size_t m0 = (size_t)blockIdx.y * 128;
  const size_t n0 = (size_t)blockIdx.x * 128;

  v8f acc[4][2];
#pragma unroll
  for (int i = 0; i < 4; i++)
#pragma unroll
    for (int j = 0; j < 2; j++) acc[i][j] = {};

  const int nkt = K >> 5;            // K / 32

  // Per-thread staging chunk coordinates (two 16B chunks per matrix).
  int rowc[2], colc[2];
  unsigned int ldsA[2][2], ldsB[2][2];   // [buf][chunk] LDS byte addresses
#pragma unroll
  for (int c = 0; c < 2; c++) {
    int id = tid + c * 256;
    rowc[c] = id >> 2;              // 4 x 16B chunks per 32-elem row
    colc[c] = (id & 3) * 8;
#pragma unroll
    for (int buf = 0; buf < 2; buf++) {
      ldsA[buf][c] = (unsigned int)(size_t)&As[buf][rowc[c] * 40 + colc[c]];
      ldsB[buf][c] = (unsigned int)(size_t)&Bs[buf][rowc[c] * 40 + colc[c]];
    }
  }

  const __bf16* Abase = A + m0 * K;
  const __bf16* Bbase = Bt + n0 * K;

  // prologue: async-stage k-tile 0 into buffer 0
#pragma unroll
  for (int c = 0; c < 2; c++) {
    async_copy_b128(ldsA[0][c], Abase + (size_t)rowc[c] * K + colc[c]);
    async_copy_b128(ldsB[0][c], Bbase + (size_t)rowc[c] * K + colc[c]);
  }

  for (int kt = 0; kt < nkt; kt++) {
    const int buf = kt & 1;
    if (kt + 1 < nkt) {
      // stage next k-tile into the other buffer, then drain previous batch
#pragma unroll
      for (int c = 0; c < 2; c++) {
        async_copy_b128(ldsA[buf ^ 1][c],
                        Abase + (size_t)rowc[c] * K + (kt + 1) * 32 + colc[c]);
        async_copy_b128(ldsB[buf ^ 1][c],
                        Bbase + (size_t)rowc[c] * K + (kt + 1) * 32 + colc[c]);
      }
      asm volatile("s_wait_asynccnt 0x4" ::: "memory");
    } else {
      asm volatile("s_wait_asynccnt 0x0" ::: "memory");
    }
    __syncthreads();   // all waves' async data for tile kt now visible

    const int khalf = (lane >> 4) * 8;
    v16bf bfr[2];
#pragma unroll
    for (int j = 0; j < 2; j++)
      bfr[j] = lds_frag(&Bs[buf][(wn * 32 + j * 16 + (lane & 15)) * 40 + khalf]);
#pragma unroll
    for (int i = 0; i < 4; i++) {
      v16bf afr = lds_frag(&As[buf][(wm * 64 + i * 16 + (lane & 15)) * 40 + khalf]);
      acc[i][0] = wmma_bf16(afr, bfr[0], acc[i][0]);
      acc[i][1] = wmma_bf16(afr, bfr[1], acc[i][1]);
    }
    __syncthreads();   // safe to overwrite buf^1 next iteration
  }

  // epilogue: C fragment layout -> memory
  const int rowoff = (lane >> 4) * 8;
  const int ncl    = lane & 15;
#pragma unroll
  for (int i = 0; i < 4; i++) {
#pragma unroll
    for (int j = 0; j < 2; j++) {
      size_t nc = n0 + wn * 32 + j * 16 + ncl;
      float bv = bias ? bias[nc] : 0.0f;
#pragma unroll
      for (int r = 0; r < 8; r++) {
        size_t mm = m0 + wm * 64 + i * 16 + rowoff + r;
        float v = acc[i][j][r] + bv;
        if (relu)  v = fmaxf(v, 0.0f);
        if (resid) v += resid[mm * N + nc];
        if (outF)  outF[mm * N + nc] = v;
        if (outB)  outB[mm * N + nc] = (__bf16)v;
      }
    }
  }
}

// ---------------------------------------------------------------------------
// Per-row L2 norms of q and k (bf16 qkv buffer, row = (b,s), head h slice)
// nq/nk layout: [(b*H + h)*S + s]
// ---------------------------------------------------------------------------
__global__ __launch_bounds__(256)
void qk_norms_kernel(const __bf16* __restrict__ qkv,
                     float* __restrict__ nq, float* __restrict__ nk) {
  int t = blockIdx.x * blockDim.x + threadIdx.x;
  if (t >= BS_ * HH) return;
  int bs = t / HH, h = t % HH;
  int b = bs / SS, s = bs % SS;
  const __bf16* qp = qkv + (size_t)bs * QKVW + h * DKK;
  float sq = 0.f, sk = 0.f;
#pragma unroll
  for (int c = 0; c < 8; c++) {
    Pack8 pq, pk;
    pq.u = *reinterpret_cast<const uint4*>(qp + c * 8);
    pk.u = *reinterpret_cast<const uint4*>(qp + 512 + c * 8);
#pragma unroll
    for (int i = 0; i < 8; i++) {
      float q = (float)pq.h[i]; sq += q * q;
      float k = (float)pk.h[i]; sk += k * k;
    }
  }
  int o = (b * HH + h) * SS + s;
  nq[o] = sqrtf(sq);
  nk[o] = sqrtf(sk);
}

// ---------------------------------------------------------------------------
// Fused cosine-sim attention (flash style, online softmax).
// Block = 128 threads (4 waves); block tile = 64 q-rows; wave tile = 16 q-rows.
// k processed in tiles of 64.  All matmuls via WMMA bf16.
// ---------------------------------------------------------------------------
__global__ __launch_bounds__(128)
void attn_kernel(const __bf16* __restrict__ qkv,
                 const float* __restrict__ nq, const float* __restrict__ nk,
                 __bf16* __restrict__ aout)
{
  __shared__ __align__(16) __bf16 Qs[64 * 72];
  __shared__ __align__(16) __bf16 Ks[64 * 72];
  __shared__ __align__(16) __bf16 Vt[64 * 72];          // transposed V: [dv][k]
  __shared__ __align__(16) __bf16 Pw[4 * 16 * 72];      // per-wave P tiles

  const int tid  = threadIdx.x;
  const int lane = tid & 31;
  const int wave = tid >> 5;

  const int nqt = SS / 64;
  int bx = blockIdx.x;
  const int qt = bx % nqt;
  const int h  = (bx / nqt) % HH;
  const int b  = bx / (nqt * HH);

  const size_t rowStride = QKVW;
  const __bf16* qbase = qkv + ((size_t)b * SS) * rowStride + h * DKK;

  // ---- stage Q tile (64 x 64) ----
#pragma unroll
  for (int c = 0; c < 4; c++) {
    int id = tid + c * 128;
    int row = id >> 3;                   // 8 uint4 per 64-elem row
    int col = (id & 7) * 8;
    *reinterpret_cast<uint4*>(&Qs[row * 72 + col]) =
        *reinterpret_cast<const uint4*>(qbase + (size_t)(qt * 64 + row) * rowStride + col);
  }
  __syncthreads();

  const int khalf = (lane >> 4) * 8;
  const int rl    = lane & 15;

  // per-wave Q fragments (whole kernel lifetime)
  v16bf aq0 = lds_frag(&Qs[(wave * 16 + rl) * 72 + 0  + khalf]);
  v16bf aq1 = lds_frag(&Qs[(wave * 16 + rl) * 72 + 32 + khalf]);

  // per-lane q norms for its 8 C-fragment rows
  float rnq[8];
#pragma unroll
  for (int r = 0; r < 8; r++) {
    int srow = qt * 64 + wave * 16 + (lane >> 4) * 8 + r;
    rnq[r] = nq[(b * HH + h) * SS + srow];
  }

  float m_run[8], l_run[8];
  v8f o_acc[4];
#pragma unroll
  for (int r = 0; r < 8; r++) { m_run[r] = -1e30f; l_run[r] = 0.0f; }
#pragma unroll
  for (int j = 0; j < 4; j++) o_acc[j] = {};

  const float* nkp = nk + (b * HH + h) * SS;

  for (int kt = 0; kt < SS / 64; kt++) {
    __syncthreads();   // protect K/Vt reuse from previous iteration's compute
    // ---- stage K tile (64 x 64, row major) ----
#pragma unroll
    for (int c = 0; c < 4; c++) {
      int id = tid + c * 128;
      int row = id >> 3;
      int col = (id & 7) * 8;
      *reinterpret_cast<uint4*>(&Ks[row * 72 + col]) =
          *reinterpret_cast<const uint4*>(qbase + 512 + (size_t)(kt * 64 + row) * rowStride + col);
    }
    // ---- stage V tile transposed: Vt[dv][k] ----
#pragma unroll
    for (int c = 0; c < 4; c++) {
      int id = tid + c * 128;
      int row = id >> 3;                 // k-row within tile
      int col = (id & 7) * 8;            // dv base
      Pack8 pv;
      pv.u = *reinterpret_cast<const uint4*>(qbase + 1024 + (size_t)(kt * 64 + row) * rowStride + col);
#pragma unroll
      for (int i = 0; i < 8; i++) Vt[(col + i) * 72 + row] = pv.h[i];
    }
    __syncthreads();

    // ---- scores S = Q * K^T for this wave's 16 rows x 64 k-cols ----
    float sv[4][8];
#pragma unroll
    for (int j = 0; j < 4; j++) {
      v16bf bk0 = lds_frag(&Ks[(j * 16 + rl) * 72 + 0  + khalf]);
      v16bf bk1 = lds_frag(&Ks[(j * 16 + rl) * 72 + 32 + khalf]);
      v8f sacc = {};
      sacc = wmma_bf16(aq0, bk0, sacc);
      sacc = wmma_bf16(aq1, bk1, sacc);
      float nkv = nkp[kt * 64 + j * 16 + rl];
#pragma unroll
      for (int r = 0; r < 8; r++) {
        float den = fmaxf(rnq[r] * nkv, 1e-8f) * 8.0f;   // TEMP = sqrt(64)
        sv[j][r] = sacc[r] / den;
      }
    }

    // ---- online softmax (row reductions across the 16-lane half) ----
#pragma unroll
    for (int r = 0; r < 8; r++) {
      float mx = fmaxf(fmaxf(sv[0][r], sv[1][r]), fmaxf(sv[2][r], sv[3][r]));
#pragma unroll
      for (int msk = 1; msk < 16; msk <<= 1)
        mx = fmaxf(mx, __shfl_xor(mx, msk, 32));
      float mnew = fmaxf(m_run[r], mx);
      float corr = __expf(m_run[r] - mnew);
      m_run[r] = mnew;
      float rsum = 0.0f;
#pragma unroll
      for (int j = 0; j < 4; j++) {
        float p = __expf(sv[j][r] - mnew);
        sv[j][r] = p;
        rsum += p;
      }
#pragma unroll
      for (int msk = 1; msk < 16; msk <<= 1)
        rsum += __shfl_xor(rsum, msk, 32);
      l_run[r] = l_run[r] * corr + rsum;
#pragma unroll
      for (int j = 0; j < 4; j++) o_acc[j][r] *= corr;
    }

    // ---- write P (bf16) to per-wave LDS in A-matrix layout source form ----
    __bf16* pw = &Pw[wave * 16 * 72];
#pragma unroll
    for (int r = 0; r < 8; r++)
#pragma unroll
      for (int j = 0; j < 4; j++)
        pw[((lane >> 4) * 8 + r) * 72 + j * 16 + rl] = (__bf16)sv[j][r];
    asm volatile("s_wait_dscnt 0" ::: "memory");   // whole-wave LDS RAW fence

    // ---- O += P * V ----
    v16bf ap0 = lds_frag(&pw[rl * 72 + 0  + khalf]);
    v16bf ap1 = lds_frag(&pw[rl * 72 + 32 + khalf]);
#pragma unroll
    for (int jj = 0; jj < 4; jj++) {
      v16bf bv0 = lds_frag(&Vt[(jj * 16 + rl) * 72 + 0  + khalf]);
      v16bf bv1 = lds_frag(&Vt[(jj * 16 + rl) * 72 + 32 + khalf]);
      o_acc[jj] = wmma_bf16(ap0, bv0, o_acc[jj]);
      o_acc[jj] = wmma_bf16(ap1, bv1, o_acc[jj]);
    }
  }

  // ---- normalize and store: aout[b, s, h*DV + dv] (bf16) ----
#pragma unroll
  for (int r = 0; r < 8; r++) {
    float inv = 1.0f / l_run[r];
    int srow = qt * 64 + wave * 16 + (lane >> 4) * 8 + r;
    size_t base = ((size_t)b * SS + srow) * (HH * DVV) + h * DVV;
#pragma unroll
    for (int jj = 0; jj < 4; jj++)
      aout[base + jj * 16 + rl] = (__bf16)(o_acc[jj][r] * inv);
  }
}

// ---------------------------------------------------------------------------
// LayerNorm over D=512: one wave per row; optional bf16 copy output.
// ---------------------------------------------------------------------------
__global__ __launch_bounds__(128)
void ln_kernel(const float* __restrict__ xin,
               const float* __restrict__ g, const float* __restrict__ bb,
               float* __restrict__ yf, __bf16* __restrict__ yb, int rows)
{
  int wave = threadIdx.x >> 5, lane = threadIdx.x & 31;
  int row = blockIdx.x * 4 + wave;
  if (row >= rows) return;
  const float* p = xin + (size_t)row * DD;
  float v[16], s = 0.f, s2 = 0.f;
#pragma unroll
  for (int i = 0; i < 16; i++) {
    v[i] = p[lane + i * 32];
    s += v[i];
    s2 += v[i] * v[i];
  }
#pragma unroll
  for (int msk = 1; msk < 32; msk <<= 1) {
    s  += __shfl_xor(s,  msk, 32);
    s2 += __shfl_xor(s2, msk, 32);
  }
  float mean = s * (1.0f / DD);
  float var  = s2 * (1.0f / DD) - mean * mean;
  float inv  = rsqrtf(var + 1e-5f);
#pragma unroll
  for (int i = 0; i < 16; i++) {
    int c = lane + i * 32;
    float y = (v[i] - mean) * inv * g[c] + bb[c];
    yf[(size_t)row * DD + c] = y;
    if (yb) yb[(size_t)row * DD + c] = (__bf16)y;
  }
}

// ---------------------------------------------------------------------------
// Host-side orchestration
// ---------------------------------------------------------------------------
static inline size_t alignup(size_t x) { return (x + 255) & ~(size_t)255; }

extern "C" void kernel_launch(void* const* d_in, const int* in_sizes, int n_in,
                              void* d_out, int out_size, void* d_ws, size_t ws_size,
                              hipStream_t stream) {
  const float* x     = (const float*)d_in[0];
  const float* w_q   = (const float*)d_in[1];
  const float* w_k   = (const float*)d_in[2];
  const float* w_v   = (const float*)d_in[3];
  const float* w_o   = (const float*)d_in[4];
  const float* w_ff1 = (const float*)d_in[5];
  const float* b_ff1 = (const float*)d_in[6];
  const float* w_ff2 = (const float*)d_in[7];
  const float* b_ff2 = (const float*)d_in[8];
  const float* g1    = (const float*)d_in[9];
  const float* b1    = (const float*)d_in[10];
  const float* g2    = (const float*)d_in[11];
  const float* b2    = (const float*)d_in[12];
  float* out = (float*)d_out;

  char* w = (char*)d_ws;
  auto take = [&](size_t bytes) { char* p = w; w += alignup(bytes); return p; };

  __bf16* xb     = (__bf16*)take((size_t)BS_ * DD * 2);
  __bf16* wqkvT  = (__bf16*)take((size_t)QKVW * DD * 2);
  __bf16* woT    = (__bf16*)take((size_t)DD * DD * 2);
  __bf16* wff1T  = (__bf16*)take((size_t)DFF_ * DD * 2);
  __bf16* wff2T  = (__bf16*)take((size_t)DD * DFF_ * 2);
  __bf16* qkv    = (__bf16*)take((size_t)BS_ * QKVW * 2);
  float*  nqv    = (float*)take((size_t)BB * HH * SS * 4);
  float*  nkv    = (float*)take((size_t)BB * HH * SS * 4);
  __bf16* aout   = (__bf16*)take((size_t)BS_ * DD * 2);
  float*  resid1 = (float*)take((size_t)BS_ * DD * 4);
  float*  h1     = (float*)take((size_t)BS_ * DD * 4);
  __bf16* h1b    = (__bf16*)take((size_t)BS_ * DD * 2);
  __bf16* ffh    = (__bf16*)take((size_t)BS_ * DFF_ * 2);

  // 1) x -> bf16
  {
    int n4 = BS_ * DD / 4;
    cvt_bf16_kernel<<<(n4 + 255) / 256, 256, 0, stream>>>(x, xb, n4);
  }
  // 2) weight transposes (f32 [KxN] -> bf16 [NxK])
  transpose_cvt_kernel<<<dim3(16, 16), 256, 0, stream>>>(w_q, wqkvT + 0,              DD, DD);
  transpose_cvt_kernel<<<dim3(16, 16), 256, 0, stream>>>(w_k, wqkvT + 512 * DD,       DD, DD);
  transpose_cvt_kernel<<<dim3(16, 16), 256, 0, stream>>>(w_v, wqkvT + 1024 * DD,      DD, DD);
  transpose_cvt_kernel<<<dim3(16, 16), 256, 0, stream>>>(w_o, woT,                    DD, DD);
  transpose_cvt_kernel<<<dim3(64, 16), 256, 0, stream>>>(w_ff1, wff1T,                DD, DFF_);
  transpose_cvt_kernel<<<dim3(16, 64), 256, 0, stream>>>(w_ff2, wff2T,                DFF_, DD);

  // 3) fused QKV projection: [16384 x 1536] = xb @ [wq|wk|wv]
  gemm_bf16_kernel<<<dim3(QKVW / 128, BS_ / 128), 256, 0, stream>>>(
      xb, wqkvT, BS_, QKVW, DD, nullptr, nullptr, nullptr, qkv, 0);

  // 4) q/k row norms
  qk_norms_kernel<<<(BS_ * HH) / 256, 256, 0, stream>>>(qkv, nqv, nkv);

  // 5) fused cosine-sim attention -> aout [b, s, h*dv] bf16
  attn_kernel<<<BB * HH * (SS / 64), 128, 0, stream>>>(qkv, nqv, nkv, aout);

  // 6) output projection + residual x -> resid1 (f32)
  gemm_bf16_kernel<<<dim3(DD / 128, BS_ / 128), 256, 0, stream>>>(
      aout, woT, BS_, DD, DD, nullptr, x, resid1, nullptr, 0);

  // 7) LN1 -> h1 (f32) + h1b (bf16)
  ln_kernel<<<BS_ / 4, 128, 0, stream>>>(resid1, g1, b1, h1, h1b, BS_);

  // 8) FFN1: relu(h1 @ w_ff1 + b_ff1) -> ffh (bf16)
  gemm_bf16_kernel<<<dim3(DFF_ / 128, BS_ / 128), 256, 0, stream>>>(
      h1b, wff1T, BS_, DFF_, DD, b_ff1, nullptr, nullptr, ffh, 1);

  // 9) FFN2: ffh @ w_ff2 + b_ff2 + h1 -> resid1 (reused, f32)
  gemm_bf16_kernel<<<dim3(DD / 128, BS_ / 128), 256, 0, stream>>>(
      ffh, wff2T, BS_, DD, DFF_, b_ff2, h1, resid1, nullptr, 0);

  // 10) LN2 -> d_out (f32)
  ln_kernel<<<BS_ / 4, 128, 0, stream>>>(resid1, g2, b2, out, nullptr, BS_);
}
